// OpponentModelOracle_31739808318169
// MI455X (gfx1250) — compile-verified, matrix operand order checked by reading, not run on hardware
//
#include <hip/hip_runtime.h>

typedef __attribute__((ext_vector_type(16))) _Float16 v16h;
typedef __attribute__((ext_vector_type(8)))  float    v8f;
typedef __attribute__((ext_vector_type(4)))  float    v4f;   // native clang vector (NT-load legal)

#define HW     4096   // 64*64 cells per image
#define TPB    256    // 8 wave32
#define CPT    16     // cells per thread
#define NWAVES 8

// Merge two (min1, idx1, min2) states. Total order: smaller dist wins,
// equal dist -> smaller row-major index wins (matches jnp.argmin / torch).
__device__ __forceinline__ void merge2(float& m1, int& i1, float& m2,
                                       float bm1, int bi1, float bm2) {
  if (bm1 < m1 || (bm1 == m1 && bi1 < i1)) {
    m2 = fminf(bm2, m1);   // old winner becomes runner-up candidate
    m1 = bm1; i1 = bi1;
  } else {
    m2 = fminf(m2, bm1);   // loser's best is runner-up candidate
  }
}

__global__ __launch_bounds__(TPB)
void oracle_kernel(const float* __restrict__ x,
                   const float* __restrict__ opp_start,
                   float* __restrict__ out) {
  const int b    = blockIdx.x;
  const int t    = threadIdx.x;
  const int wid  = t >> 5;
  const int lane = t & 31;
  const float* __restrict__ base = x + (size_t)b * (HW * 8);

  __shared__ int   s_food[NWAVES], s_opp[NWAVES], s_fopp[NWAVES];
  __shared__ float s_m1[NWAVES],  s_m2[NWAVES];
  __shared__ int   s_i1[NWAVES];

  // ---- Phase 1: single streaming pass; keep flags as register bitmasks ----
  unsigned foodmask = 0u, oppmask = 0u;
  #pragma unroll
  for (int k = 0; k < CPT; ++k) {
    const int cell = k * TPB + t;                         // coalesced across lanes
    const v4f* p = (const v4f*)(base + (size_t)cell * 8); // ch0..ch3 in one b128
    const v4f v = __builtin_nontemporal_load(p);          // NT: zero reuse, spare L2
    foodmask |= (v.y == 1.0f) ? (1u << k) : 0u;           // channel 1
    oppmask  |= (v.w == 1.0f) ? (1u << k) : 0u;           // channel 3
  }

  // ---- Count reduction on the matrix pipe (v_wmma) ----
  // A tile = 16 f16 flags per lane (full 16x32 A matrix), B = ones:
  // every D column equals the row-sum vector, so sum(all D)/16 = sum(flags).
  // Exact: f16 {0,1} products accumulated in f32, counts <= 512 per wave.
  // Flag->f16 via ternary on f16 constants => v_cndmask_b16 (no cvt chain).
  const _Float16 h1 = (_Float16)1.0f;
  const _Float16 h0 = (_Float16)0.0f;
  v16h af, ao, ones;
  #pragma unroll
  for (int k = 0; k < 16; ++k) {
    af[k]   = ((foodmask >> k) & 1u) ? h1 : h0;
    ao[k]   = ((oppmask  >> k) & 1u) ? h1 : h0;
    ones[k] = h1;
  }
  v8f zc = {};
  v8f df = __builtin_amdgcn_wmma_f32_16x16x32_f16(false, af, false, ones,
                                                  (short)0, zc, false, false);
  v8f dp = __builtin_amdgcn_wmma_f32_16x16x32_f16(false, ao, false, ones,
                                                  (short)0, zc, false, false);
  float fs = 0.0f, os = 0.0f;
  #pragma unroll
  for (int j = 0; j < 8; ++j) { fs += df[j]; os += dp[j]; }

  // first opponent cell (row-major) owned by this thread
  int fo = oppmask ? ((__ffs(oppmask) - 1) * TPB + t) : 0x7FFFFFFF;

  // wave32 butterfly reduce: counts (sum) + first_opp (min)
  #pragma unroll
  for (int m = 16; m >= 1; m >>= 1) {
    fs += __shfl_xor(fs, m, 32);
    os += __shfl_xor(os, m, 32);
    const int o = __shfl_xor(fo, m, 32);
    fo = (o < fo) ? o : fo;
  }
  if (lane == 0) {
    s_food[wid] = (int)(fs * 0.0625f + 0.5f);  // /16: D duplicates row-sums 16x
    s_opp[wid]  = (int)(os * 0.0625f + 0.5f);
    s_fopp[wid] = fo;
  }
  __syncthreads();

  int n_food = 0, n_opp = 0, first_opp = 0x7FFFFFFF;
  #pragma unroll
  for (int w = 0; w < NWAVES; ++w) {
    n_food   += s_food[w];
    n_opp    += s_opp[w];
    first_opp = min(first_opp, s_fopp[w]);
  }
  if (n_opp == 0) first_opp = 0;                 // argmax of all-False == 0
  const float orow = (float)(first_opp >> 6);
  const float ocol = (float)(first_opp & 63);

  // ---- Phase 2: two smallest food->opponent distances (no memory reread) ----
  const float INF = __builtin_inff();
  float m1 = INF, m2 = INF; int i1 = 0x7FFFFFFF;
  #pragma unroll
  for (int k = 0; k < CPT; ++k) {
    if ((foodmask >> k) & 1u) {
      const int cell = k * TPB + t;
      const float dr = (float)(cell >> 6) - orow;
      const float dc = (float)(cell & 63) - ocol;
      const float d  = sqrtf(dr * dr + dc * dc);
      if (d < m1 || (d == m1 && cell < i1)) { m2 = m1; m1 = d; i1 = cell; }
      else                                  { m2 = fminf(m2, d); }
    }
  }
  #pragma unroll
  for (int m = 16; m >= 1; m >>= 1) {
    const float bm1 = __shfl_xor(m1, m, 32);
    const int   bi1 = __shfl_xor(i1, m, 32);
    const float bm2 = __shfl_xor(m2, m, 32);
    merge2(m1, i1, m2, bm1, bi1, bm2);
  }
  if (lane == 0) { s_m1[wid] = m1; s_i1[wid] = i1; s_m2[wid] = m2; }
  __syncthreads();

  float g1 = s_m1[0]; int gi = s_i1[0]; float g2 = s_m2[0];
  #pragma unroll
  for (int w = 1; w < NWAVES; ++w) merge2(g1, gi, g2, s_m1[w], s_i1[w], s_m2[w]);

  // ---- Branch flags (scalar per image, uniform across block) ----
  const float diff = g2 - g1;                       // finite whenever branchA
  const bool matches_start = (orow == opp_start[0]) && (ocol == opp_start[1]);
  const bool branchA       = (n_food > 1) && (n_opp > 0) && !matches_start;
  const bool ambiguous     = (branchA && (diff < 0.1f)) || ((n_food > 1) && !branchA);
  const bool pick_nearest  = (branchA && (diff >= 0.1f)) || (n_food == 1);

  // ---- Phase 3: coalesced NT output stores ----
  float* __restrict__ ob = out + (size_t)b * HW;
  #pragma unroll
  for (int k = 0; k < CPT; ++k) {
    const int cell = k * TPB + t;
    float hit;
    if (ambiguous)          hit = (float)((foodmask >> k) & 1u);
    else if (pick_nearest)  hit = (cell == gi) ? 1.0f : 0.0f;
    else                    hit = 0.0f;
    __builtin_nontemporal_store(-10.0f + 20.0f * hit, ob + cell);
  }
}

extern "C" void kernel_launch(void* const* d_in, const int* in_sizes, int n_in,
                              void* d_out, int out_size, void* d_ws, size_t ws_size,
                              hipStream_t stream) {
  (void)n_in; (void)out_size; (void)d_ws; (void)ws_size;
  const float* x         = (const float*)d_in[0];
  const float* opp_start = (const float*)d_in[1];
  float* out             = (float*)d_out;
  const int B = in_sizes[0] / (HW * 8);     // 4096 for the reference shapes
  oracle_kernel<<<dim3(B), dim3(TPB), 0, stream>>>(x, opp_start, out);
}
